// TypeAreaMultiHeadAttention_36507222016312
// MI455X (gfx1250) — matrix operand order repaired and checked by reading, not compiled, not run on hardware
//
#include <hip/hip_runtime.h>

// MI455X / gfx1250, wave32, WMMA 16x16x32 f16 -> f32, TDM K-tile staging
typedef __attribute__((ext_vector_type(16))) _Float16 v16h;
typedef __attribute__((ext_vector_type(8)))  _Float16 v8h;
typedef __attribute__((ext_vector_type(8)))  float    v8f;
typedef __attribute__((ext_vector_type(2)))  __fp16   g2h;
typedef __attribute__((ext_vector_type(4)))  __fp16   g4h;
typedef __attribute__((ext_vector_type(8)))  __fp16   g8h;
typedef __attribute__((ext_vector_type(4)))  unsigned int u32x4;
typedef __attribute__((ext_vector_type(4)))  int      i32x4;
typedef __attribute__((ext_vector_type(8)))  int      i32x8;

#define BB 8
#define LL 1024
#define DM 128
#define HH 8
#define HD 1024
#define MROWS (BB*LL)   // 8192

#if __has_builtin(__builtin_amdgcn_tensor_load_to_lds) && __has_builtin(__builtin_amdgcn_s_wait_tensorcnt)
#define HAVE_TDM 1
#else
#define HAVE_TDM 0
#endif

__device__ __forceinline__ v16h cat8(v8h lo, v8h hi) {
  return __builtin_shufflevector(lo, hi, 0,1,2,3,4,5,6,7,8,9,10,11,12,13,14,15);
}

// 8x f32 -> 8x f16 with packed converts (v_cvt_pk_f16_f32)
__device__ __forceinline__ v8h cvt8(float4 a, float4 b) {
  g2h p0 = __builtin_amdgcn_cvt_pkrtz(a.x, a.y);
  g2h p1 = __builtin_amdgcn_cvt_pkrtz(a.z, a.w);
  g2h p2 = __builtin_amdgcn_cvt_pkrtz(b.x, b.y);
  g2h p3 = __builtin_amdgcn_cvt_pkrtz(b.z, b.w);
  g4h q0 = __builtin_shufflevector(p0, p1, 0, 1, 2, 3);
  g4h q1 = __builtin_shufflevector(p2, p3, 0, 1, 2, 3);
  g8h r  = __builtin_shufflevector(q0, q1, 0, 1, 2, 3, 4, 5, 6, 7);
  return __builtin_bit_cast(v8h, r);
}

__device__ __forceinline__ v8f wmma16(v16h a, v16h b, v8f c) {
  return __builtin_amdgcn_wmma_f32_16x16x32_f16(false, a, false, b, (short)0, c, false, false);
}

// ---------------------------------------------------------------------------
// Kernel 1: combined projections.  Qc = qa@Wqa + qs@Wqs (same for K, V).
// Output row-major (B*L, H*D) f16 == (B,H,L,D) reinterpretation for attention.
// grid = (HD/64, MROWS/64, 3), block = 256 (8 waves, 4x2 wave tiles of 16x32)
// ---------------------------------------------------------------------------
__global__ __launch_bounds__(256) void proj_kernel(
    const float* __restrict__ qa, const float* __restrict__ ka, const float* __restrict__ va,
    const float* __restrict__ qs, const float* __restrict__ ks, const float* __restrict__ vs,
    const float* __restrict__ Wqa, const float* __restrict__ Wka, const float* __restrict__ Wva,
    const float* __restrict__ Wqs, const float* __restrict__ Wks, const float* __restrict__ Wvs,
    _Float16* __restrict__ Qc, _Float16* __restrict__ Kc, _Float16* __restrict__ Vc)
{
  __shared__ __align__(16) _Float16 lA[64 * 40];   // 64 rows x 32 k (stride 40)
  __shared__ __align__(16) _Float16 lB[64 * 40];   // 64 n    x 32 k (transposed W)

  const int p    = blockIdx.z;
  const int tid  = threadIdx.x;
  const int lane = tid & 31;
  const int w    = tid >> 5;
  const int wm   = w & 3;
  const int wn   = w >> 2;
  const int m0   = blockIdx.y * 64;
  const int n0   = blockIdx.x * 64;
  const int m    = lane & 15;
  const bool lo16 = (lane < 16);

  const float *X0, *X1, *W0, *W1;
  _Float16* outp;
  if (p == 0)      { X0 = qa; X1 = qs; W0 = Wqa; W1 = Wqs; outp = Qc; }
  else if (p == 1) { X0 = ka; X1 = ks; W0 = Wka; W1 = Wks; outp = Kc; }
  else             { X0 = va; X1 = vs; W0 = Wva; W1 = Wvs; outp = Vc; }

  // staging coordinates (per thread: 8 contiguous elements)
  const int ar = tid >> 2, ac = (tid & 3) * 8;      // A: 64 rows x 32 k
  const int bk = tid >> 3, bn = (tid & 7) * 8;      // B: 32 k x 64 n

  v8f acc0 = 0.0f, acc1 = 0.0f;

  for (int pass = 0; pass < 2; ++pass) {
    const float* X = pass ? X1 : X0;
    const float* W = pass ? W1 : W0;
    #pragma unroll
    for (int kc = 0; kc < 4; ++kc) {
      const int k0 = kc * 32;
      {  // stage A tile row-major, vectorized
        const float* src = &X[(size_t)(m0 + ar) * DM + k0 + ac];
        float4 a0 = *(const float4*)src;
        float4 a1 = *(const float4*)(src + 4);
        *(v8h*)&lA[ar * 40 + ac] = cvt8(a0, a1);
      }
      {  // stage B tile transposed: lB[n][k] = W[k0+k][n0+n]
        const float* src = &W[(size_t)(k0 + bk) * HD + n0 + bn];
        float4 b0 = *(const float4*)src;
        float4 b1 = *(const float4*)(src + 4);
        v8h h8 = cvt8(b0, b1);
        #pragma unroll
        for (int j = 0; j < 8; ++j) lB[(bn + j) * 40 + bk] = h8[j];
      }
      __syncthreads();

      const int mr = wm * 16 + m;
      const int kb = lo16 ? 0 : 16;
      v16h a = cat8(*(const v8h*)&lA[mr * 40 + kb],
                    *(const v8h*)&lA[mr * 40 + kb + 8]);

      const int kb2 = lo16 ? 0 : 16;
      {
        int nl = wn * 32 + m;
        acc0 = wmma16(a, cat8(*(const v8h*)&lB[nl * 40 + kb2],
                              *(const v8h*)&lB[nl * 40 + kb2 + 8]), acc0);
      }
      {
        int nl = wn * 32 + 16 + m;
        acc1 = wmma16(a, cat8(*(const v8h*)&lB[nl * 40 + kb2],
                              *(const v8h*)&lB[nl * 40 + kb2 + 8]), acc1);
      }
      __syncthreads();
    }
  }

  #pragma unroll
  for (int v = 0; v < 8; ++v) {
    int row = m0 + wm * 16 + (lo16 ? v : v + 8);
    int c0  = n0 + wn * 32 + m;
    outp[(size_t)row * HD + c0]      = (_Float16)acc0[v];
    outp[(size_t)row * HD + c0 + 16] = (_Float16)acc1[v];
  }
}

// ---------------------------------------------------------------------------
// Kernel 2: flash attention per (b,h).  Q,K,V are contiguous (L,128) f16 slabs.
// K block (32x128, no transpose needed) staged into LDS by the Tensor Data
// Mover (pad: 64-DWORD interval + 4 DWORDs -> row stride 136 halves).
// grid = (L/128, B*H), block = 256.
// ---------------------------------------------------------------------------
__global__ __launch_bounds__(256) void attn_kernel(
    const _Float16* __restrict__ Qc, const _Float16* __restrict__ Kc,
    const _Float16* __restrict__ Vc, _Float16* __restrict__ Of)
{
  __shared__ __align__(16) _Float16 lK[32 * 136];     // K:  key(32) x d(128), stride 136
  __shared__ __align__(16) _Float16 lV[128 * 40];     // Vt: d(128)  x key(32), stride 40
  __shared__ __align__(16) _Float16 lP[8][16 * 40];   // per-wave P tile 16x32

  const int tid  = threadIdx.x;
  const int lane = tid & 31;
  const int w    = tid >> 5;
  const int bh   = blockIdx.y;
  const int b    = bh >> 3, h = bh & 7;
  const size_t base = (size_t)bh * LL * DM;
  const int l0   = blockIdx.x * 128;
  const int m    = lane & 15;
  const bool lo16 = (lane < 16);
  const float rscale = 0.04419417382415922f;  // 1/sqrt(4*D)

  // Q A-vectors for all four 32-wide d-chunks (live in VGPRs whole kernel)
  v16h aq[4];
  {
    const _Float16* qrow = Qc + base + (size_t)(l0 + w * 16 + m) * DM;
    const int kb = lo16 ? 0 : 16;
    #pragma unroll
    for (int c = 0; c < 4; ++c)
      aq[c] = cat8(*(const v8h*)(qrow + c * 32 + kb),
                   *(const v8h*)(qrow + c * 32 + kb + 8));
  }

  v8f o[8];
  #pragma unroll
  for (int i = 0; i < 8; ++i) o[i] = 0.0f;
  float mrow[8], lrow[8];
  #pragma unroll
  for (int v = 0; v < 8; ++v) { mrow[v] = -1e30f; lrow[v] = 0.0f; }

  for (int j0 = 0; j0 < LL; j0 += 32) {
#if HAVE_TDM
    if (w == 0) {  // wave-uniform: one TDM descriptor per key block
      unsigned ldsOff = (unsigned)(size_t)&lK[0];        // LDS aperture: addr[31:0]
      unsigned long long ga =
          (unsigned long long)(size_t)(Kc + base + (size_t)j0 * DM);
      u32x4 g0;
      g0[0] = 1u;                                        // count = 1 (valid D#)
      g0[1] = ldsOff;                                    // lds_addr
      g0[2] = (unsigned)(ga & 0xFFFFFFFFu);              // global_addr[31:0]
      g0[3] = (unsigned)((ga >> 32) & 0x1FFFFFFu) | (2u << 30);  // addr[56:32], type=2
      i32x8 g1;
      g1[0] = (1 << 16) | (1 << 20) | (5 << 22) | (3 << 25);
      //        data=2B    pad_en     ivl=64DW    amt=4DW  -> LDS row stride 136 halves
      g1[1] = (int)(128u << 16);   // tensor_dim0 = 128 (d)
      g1[2] = (int)(1024u << 16);  // tensor_dim1 = 1024 (keys)
      g1[3] = (int)(128u << 16);   // tile_dim0 = 128
      g1[4] = 32;                  // tile_dim1 = 32
      g1[5] = 128;                 // tensor_dim0_stride = 128
      g1[6] = 0;
      g1[7] = 0;
      i32x4 z4 = {0, 0, 0, 0};
#if __clang_major__ >= 23
      i32x8 z8 = {0, 0, 0, 0, 0, 0, 0, 0};
      __builtin_amdgcn_tensor_load_to_lds(g0, g1, z4, z4, z8, 0);
#else
      __builtin_amdgcn_tensor_load_to_lds(g0, g1, z4, z4, 0);
#endif
      __builtin_amdgcn_s_wait_tensorcnt(0);
    }
#else
    #pragma unroll
    for (int i = 0; i < 2; ++i) {  // fallback: vectorized copy
      int q = tid + i * 256;
      int key = q >> 4, d0 = (q & 15) * 8;
      *(v8h*)&lK[key * 136 + d0] =
          *(const v8h*)(Kc + base + (size_t)(j0 + key) * DM + d0);
    }
#endif
    // stage V transposed: lV[d][key] = V[j0+key][d]  (b128 loads, b16 scatter)
    #pragma unroll
    for (int i = 0; i < 2; ++i) {
      int q = tid + i * 256;
      int key = q >> 4, d0 = (q & 15) * 8;
      v8h vv = *(const v8h*)(Vc + base + (size_t)(j0 + key) * DM + d0);
      #pragma unroll
      for (int j = 0; j < 8; ++j) lV[(d0 + j) * 40 + key] = vv[j];
    }
    if (j0 + 32 < LL)  // global_prefetch_b8 of next V block
      __builtin_prefetch(&Vc[base + (size_t)(j0 + 32) * DM + tid * 16], 0, 1);
    __syncthreads();

    // S = Q @ K^T  (reduction over d; lK rows are d-contiguous -> B layout)
    v8f s0 = 0.0f, s1 = 0.0f;
    #pragma unroll
    for (int c = 0; c < 4; ++c) {
      const int dk = c * 32 + (lo16 ? 0 : 16);
      const _Float16* kr0 = &lK[m * 136 + dk];
      s0 = wmma16(aq[c], cat8(*(const v8h*)kr0, *(const v8h*)(kr0 + 8)), s0);
      const _Float16* kr1 = &lK[(16 + m) * 136 + dk];
      s1 = wmma16(aq[c], cat8(*(const v8h*)kr1, *(const v8h*)(kr1 + 8)), s1);
    }

    // online softmax (per-row stats, rows live across 16 lanes of half-wave)
    #pragma unroll
    for (int v = 0; v < 8; ++v) {
      float a0 = s0[v] * rscale;
      float a1 = s1[v] * rscale;
      float t = fmaxf(a0, a1);
      t = fmaxf(t, __shfl_xor(t, 1, 32));
      t = fmaxf(t, __shfl_xor(t, 2, 32));
      t = fmaxf(t, __shfl_xor(t, 4, 32));
      t = fmaxf(t, __shfl_xor(t, 8, 32));
      float mnew  = fmaxf(mrow[v], t);
      float alpha = __expf(mrow[v] - mnew);
      mrow[v] = mnew;
      float p0 = __expf(a0 - mnew);
      float p1 = __expf(a1 - mnew);
      float ps = p0 + p1;
      ps += __shfl_xor(ps, 1, 32);
      ps += __shfl_xor(ps, 2, 32);
      ps += __shfl_xor(ps, 4, 32);
      ps += __shfl_xor(ps, 8, 32);
      lrow[v] = lrow[v] * alpha + ps;
      #pragma unroll
      for (int td = 0; td < 8; ++td) o[td][v] *= alpha;
      int r = lo16 ? v : v + 8;
      lP[w][r * 40 + m]      = (_Float16)p0;
      lP[w][r * 40 + 16 + m] = (_Float16)p1;
    }

    // re-read P in A layout (same-wave LDS round trip, DS ops are in order)
    const int kb = lo16 ? 0 : 16;
    v16h ap = cat8(*(const v8h*)&lP[w][m * 40 + kb],
                   *(const v8h*)&lP[w][m * 40 + kb + 8]);

    // O += P @ V  (reduction over key; lV is key-contiguous -> B layout)
    #pragma unroll
    for (int td = 0; td < 8; ++td) {
      const _Float16* vr = &lV[(td * 16 + m) * 40 + (lo16 ? 0 : 16)];
      o[td] = wmma16(ap, cat8(*(const v8h*)vr, *(const v8h*)(vr + 8)), o[td]);
    }
    __syncthreads();
  }

  // epilogue: O/l, scatter to (b, l, h*128+d) layout for the final GEMM
  const size_t obase = ((size_t)(b * LL) + l0 + w * 16) * HD + h * DM;
  #pragma unroll
  for (int v = 0; v < 8; ++v) {
    int r = lo16 ? v : v + 8;
    float inv = 1.0f / lrow[v];
    #pragma unroll
    for (int td = 0; td < 8; ++td)
      Of[obase + (size_t)r * HD + td * 16 + m] = (_Float16)(o[td][v] * inv);
  }
}

// ---------------------------------------------------------------------------
// Kernel 3: y = Of @ W_final + q_a + q_s, then LayerNorm over D=128.
// grid = MROWS/32, block = 256 (8 waves: 2 row-tiles x 4 col-tiles of 32)
// ---------------------------------------------------------------------------
__global__ __launch_bounds__(256) void final_kernel(
    const _Float16* __restrict__ Of, const float* __restrict__ Wf,
    const float* __restrict__ qa, const float* __restrict__ qs,
    const float* __restrict__ gamma, const float* __restrict__ beta,
    float* __restrict__ out)
{
  __shared__ __align__(16) _Float16 lB[128 * 40];  // Wf chunk transposed
  __shared__ float ysh[32 * 132];

  const int tid  = threadIdx.x;
  const int lane = tid & 31;
  const int w    = tid >> 5;
  const int wm   = w & 1;
  const int wn   = w >> 1;
  const int m0   = blockIdx.x * 32;
  const int m    = lane & 15;
  const bool lo16 = (lane < 16);
  const int bk = tid >> 3, bn = (tid & 7) * 16;    // staging coords

  v8f acc0 = 0.0f, acc1 = 0.0f;
  const _Float16* arow = Of + (size_t)(m0 + wm * 16 + m) * HD;

  for (int kc = 0; kc < 32; ++kc) {
    const int k0 = kc * 32;
    {  // stage Wf chunk transposed: lB[n][k] = Wf[k0+k][n]
      const float* src = &Wf[(size_t)(k0 + bk) * DM + bn];
      v8h h0 = cvt8(*(const float4*)src,       *(const float4*)(src + 4));
      v8h h1 = cvt8(*(const float4*)(src + 8), *(const float4*)(src + 12));
      #pragma unroll
      for (int j = 0; j < 8; ++j) {
        lB[(bn + j) * 40 + bk]     = h0[j];
        lB[(bn + 8 + j) * 40 + bk] = h1[j];
      }
    }
    __syncthreads();

    const int kb = k0 + (lo16 ? 0 : 16);
    v16h a = cat8(*(const v8h*)(arow + kb), *(const v8h*)(arow + kb + 8));
    const int kb2 = lo16 ? 0 : 16;
    {
      int nl = wn * 32 + m;
      acc0 = wmma16(a, cat8(*(const v8h*)&lB[nl * 40 + kb2],
                            *(const v8h*)&lB[nl * 40 + kb2 + 8]), acc0);
    }
    {
      int nl = wn * 32 + 16 + m;
      acc1 = wmma16(a, cat8(*(const v8h*)&lB[nl * 40 + kb2],
                            *(const v8h*)&lB[nl * 40 + kb2 + 8]), acc1);
    }
    __syncthreads();
  }

  #pragma unroll
  for (int v = 0; v < 8; ++v) {
    int r = wm * 16 + (lo16 ? v : v + 8);
    int c = wn * 32 + m;
    ysh[r * 132 + c]      = acc0[v];
    ysh[r * 132 + c + 16] = acc1[v];
  }
  __syncthreads();

  // LayerNorm: 8 lanes per row, 16 cols each
  const int row = tid >> 3;
  const int ci  = (tid & 7) * 16;
  const size_t g = (size_t)(m0 + row) * DM;
  float x[16], s = 0.0f, ss = 0.0f;
  #pragma unroll
  for (int j = 0; j < 16; ++j) {
    float t = ysh[row * 132 + ci + j] + qa[g + ci + j] + qs[g + ci + j];
    x[j] = t; s += t; ss += t * t;
  }
  s += __shfl_xor(s, 1, 32);  ss += __shfl_xor(ss, 1, 32);
  s += __shfl_xor(s, 2, 32);  ss += __shfl_xor(ss, 2, 32);
  s += __shfl_xor(s, 4, 32);  ss += __shfl_xor(ss, 4, 32);
  const float mean = s * (1.0f / 128.0f);
  const float var  = ss * (1.0f / 128.0f) - mean * mean;
  const float rstd = rsqrtf(var + 1e-6f);
  #pragma unroll
  for (int j = 0; j < 16; ++j) {
    int col = ci + j;
    out[g + col] = (x[j] - mean) * rstd * gamma[col] + beta[col];
  }
}

// ---------------------------------------------------------------------------
extern "C" void kernel_launch(void* const* d_in, const int* in_sizes, int n_in,
                              void* d_out, int out_size, void* d_ws, size_t ws_size,
                              hipStream_t stream) {
  const float* qa  = (const float*)d_in[0];
  const float* ka  = (const float*)d_in[1];
  const float* va  = (const float*)d_in[2];
  const float* qs  = (const float*)d_in[3];
  const float* ksv = (const float*)d_in[4];
  const float* vsv = (const float*)d_in[5];
  const float* Wqa = (const float*)d_in[6];
  const float* Wka = (const float*)d_in[7];
  const float* Wva = (const float*)d_in[8];
  const float* Wqs = (const float*)d_in[9];
  const float* Wks = (const float*)d_in[10];
  const float* Wvs = (const float*)d_in[11];
  const float* Wf  = (const float*)d_in[12];
  const float* gam = (const float*)d_in[13];
  const float* bet = (const float*)d_in[14];

  const size_t NE = (size_t)MROWS * HD;  // 8,388,608 elems per f16 buffer
  _Float16* Qc = (_Float16*)d_ws;
  _Float16* Kc = Qc + NE;
  _Float16* Vc = Kc + NE;
  _Float16* Of = Vc + NE;

  dim3 g1(HD / 64, MROWS / 64, 3);
  proj_kernel<<<g1, 256, 0, stream>>>(qa, ka, va, qs, ksv, vsv,
                                      Wqa, Wka, Wva, Wqs, Wks, Wvs, Qc, Kc, Vc);
  dim3 g2(LL / 128, BB * HH);
  attn_kernel<<<g2, 256, 0, stream>>>(Qc, Kc, Vc, Of);
  final_kernel<<<MROWS / 32, 256, 0, stream>>>(Of, Wf, qa, qs, gam, bet,
                                               (float*)d_out);
}
